// VimBlock_71640054497754
// MI455X (gfx1250) — compile-verified, matrix operand order checked by reading, not run on hardware
//
#include <hip/hip_runtime.h>
#include <hip/hip_bf16.h>
#include <math.h>

// Problem dims (fixed by the reference)
#define BB 4
#define MM 4096
#define MMP (MM + 2)          // padded rows per batch (1 zero halo row each side)
#define DD 768
#define EE 1536
#define NN 16
#define KK 3

typedef _Float16 v16h __attribute__((ext_vector_type(16)));
typedef _Float16 v8h  __attribute__((ext_vector_type(8)));
typedef float    v8f  __attribute__((ext_vector_type(8)));

__device__ __forceinline__ v8f zero_v8f() {
  v8f v;
#pragma unroll
  for (int i = 0; i < 8; ++i) v[i] = 0.f;
  return v;
}

// Per-lane WMMA fragment: two contiguous 16B chunks at p and p+16 halves.
// CDNA5 16-bit A 16x32 layout: lane<16 holds K 0..7 & 16..23 of its row,
// lane>=16 holds K 8..15 & 24..31 (caller bakes sel*8 into p).
__device__ __forceinline__ v16h load_frag16(const _Float16* __restrict__ p) {
  v8h a = *(const v8h*)p;
  v8h b = *(const v8h*)(p + 16);
  v16h r;
#pragma unroll
  for (int i = 0; i < 8; ++i) { r[i] = a[i]; r[i + 8] = b[i]; }
  return r;
}

enum { ACT_NONE = 0, ACT_SILU = 1, ACT_SOFTPLUS = 2 };

// ---------------------------------------------------------------------------
// Generic WMMA GEMM: out[M][N] = act(A[M][K] @ W[N][K]^T + bias + bias2) (+res)
// Each wave computes a 16x(16*NT) tile: one A fragment feeds NT WMMAs.
// Ping-pong double-buffered K loop (2x unrolled): fragments alternate between
// two buffer sets so no cross-iteration register copies are needed; loads for
// the next 32-wide K step are always in flight under the current WMMAs.
// Requires Kdim/32 even (holds for 768 and 1536).
// PAD_OUT remaps output rows into the [B][M+2][E] zero-halo conv layout.
// ---------------------------------------------------------------------------
template <int NT, int ACT, bool STORE_H, bool ADD_RES, bool PAD_OUT>
__global__ __launch_bounds__(128)
void wmma_gemm_kernel(const _Float16* __restrict__ A, int lda,
                      const _Float16* __restrict__ W, int ldw,
                      const float* __restrict__ bias,
                      const float* __restrict__ bias2,
                      const float* __restrict__ res,
                      float* __restrict__ outF,
                      _Float16* __restrict__ outH,
                      int ldo, int Kdim)
{
  const int lane = threadIdx.x & 31;
  const int wave = threadIdx.x >> 5;
  const int m0 = (blockIdx.y * 4 + wave) * 16;
  const int n0 = blockIdx.x * (16 * NT);
  const int sel = lane >> 4;
  const int rA = m0 + (lane & 15);
  const int cB0 = n0 + (lane & 15);
  const _Float16* pa = A + (size_t)rA * lda + sel * 8;
  const _Float16* pb[NT];
#pragma unroll
  for (int t = 0; t < NT; ++t)
    pb[t] = W + (size_t)(cB0 + t * 16) * ldw + sel * 8;

  v8f acc[NT];
#pragma unroll
  for (int t = 0; t < NT; ++t) acc[t] = zero_v8f();

  const int S = Kdim >> 5;              // # of 32-wide K steps (even, >= 2)

  // prologue: step 0 -> buffer set A
  v16h afA = load_frag16(pa);
  v16h bfA[NT];
#pragma unroll
  for (int t = 0; t < NT; ++t) bfA[t] = load_frag16(pb[t]);
  v16h afB;
  v16h bfB[NT];

  int k = 32;
  for (int s = 0; s < S - 2; s += 2) {
    // load step s+1 -> B, compute step s from A
    afB = load_frag16(pa + k);
#pragma unroll
    for (int t = 0; t < NT; ++t) bfB[t] = load_frag16(pb[t] + k);
    __builtin_prefetch(pa + k + 256, 0, 1);     // global_prefetch_b8
#pragma unroll
    for (int t = 0; t < NT; ++t)
      acc[t] = __builtin_amdgcn_wmma_f32_16x16x32_f16(false, afA, false, bfA[t],
                                                      (short)0, acc[t],
                                                      false, false);
    k += 32;
    // load step s+2 -> A, compute step s+1 from B
    afA = load_frag16(pa + k);
#pragma unroll
    for (int t = 0; t < NT; ++t) bfA[t] = load_frag16(pb[t] + k);
#pragma unroll
    for (int t = 0; t < NT; ++t)
      acc[t] = __builtin_amdgcn_wmma_f32_16x16x32_f16(false, afB, false, bfB[t],
                                                      (short)0, acc[t],
                                                      false, false);
    k += 32;
  }
  // epilogue: load step S-1 -> B, drain A then B
  afB = load_frag16(pa + k);
#pragma unroll
  for (int t = 0; t < NT; ++t) bfB[t] = load_frag16(pb[t] + k);
#pragma unroll
  for (int t = 0; t < NT; ++t)
    acc[t] = __builtin_amdgcn_wmma_f32_16x16x32_f16(false, afA, false, bfA[t],
                                                    (short)0, acc[t],
                                                    false, false);
#pragma unroll
  for (int t = 0; t < NT; ++t)
    acc[t] = __builtin_amdgcn_wmma_f32_16x16x32_f16(false, afB, false, bfB[t],
                                                    (short)0, acc[t],
                                                    false, false);

#pragma unroll
  for (int t = 0; t < NT; ++t) {
    const int cB = cB0 + t * 16;
    float bv = bias ? bias[cB] : 0.f;
    if (bias2) bv += bias2[cB];
#pragma unroll
    for (int j = 0; j < 8; ++j) {
      int row = m0 + sel * 8 + j;               // C/D layout: VGPR j -> M=sel*8+j
      float v = acc[t][j] + bv;
      if (ACT == ACT_SILU)     v = v / (1.f + __expf(-v));
      if (ACT == ACT_SOFTPLUS) v = (v > 20.f) ? v : log1pf(__expf(v));
      int prow = PAD_OUT ? (row + (row / MM) * 2 + 1) : row;
      size_t idx = (size_t)prow * ldo + cB;
      if (ADD_RES) v += res[idx];
      if (STORE_H) outH[idx] = (_Float16)v;
      else         outF[idx] = v;
    }
  }
}

// ---------------------------------------------------------------------------
// Conv1d(E,E,3) + SiLU as 3 shifted WMMA GEMMs over the zero-halo input
// layout [B][M+2][E] -- no masking, no EXEC manipulation in the hot loop.
// Weights pre-transposed to [k][oc][ic]. 16x64 wave tiles, ping-pong
// double-buffered K loop per tap (EE/32 = 48 steps, even).
// ---------------------------------------------------------------------------
__global__ __launch_bounds__(128)
void wmma_conv_silu_kernel(const _Float16* __restrict__ Xpad, // [B*(M+2)][E]
                           const _Float16* __restrict__ Wt,   // [3][E][E]
                           const float* __restrict__ bias,    // [E]
                           _Float16* __restrict__ out)        // [B*M][E]
{
  const int lane = threadIdx.x & 31;
  const int wave = threadIdx.x >> 5;
  const int m0 = (blockIdx.y * 4 + wave) * 16;
  const int n0 = blockIdx.x * 64;
  const int sel = lane >> 4;
  const int rA = m0 + (lane & 15);            // global row in [0, B*M)
  const int b  = rA / MM;
  const int ml = rA - b * MM;
  const int pr = b * MMP + ml;                // padded row of (m-1) neighbor
  const int cB0 = n0 + (lane & 15);

  v8f acc[4];
#pragma unroll
  for (int t = 0; t < 4; ++t) acc[t] = zero_v8f();

#pragma unroll
  for (int kk = 0; kk < KK; ++kk) {
    const _Float16* pa = Xpad + (size_t)(pr + kk) * EE + sel * 8;
    const _Float16* pw[4];
#pragma unroll
    for (int t = 0; t < 4; ++t)
      pw[t] = Wt + ((size_t)kk * EE + cB0 + t * 16) * EE + sel * 8;

    v16h afA = load_frag16(pa);
    v16h bfA[4];
#pragma unroll
    for (int t = 0; t < 4; ++t) bfA[t] = load_frag16(pw[t]);
    v16h afB;
    v16h bfB[4];

    int k = 32;
    for (int s = 0; s < (EE / 32) - 2; s += 2) {
      afB = load_frag16(pa + k);
#pragma unroll
      for (int t = 0; t < 4; ++t) bfB[t] = load_frag16(pw[t] + k);
      __builtin_prefetch(pa + k + 256, 0, 1);
#pragma unroll
      for (int t = 0; t < 4; ++t)
        acc[t] = __builtin_amdgcn_wmma_f32_16x16x32_f16(false, afA, false, bfA[t],
                                                        (short)0, acc[t],
                                                        false, false);
      k += 32;
      afA = load_frag16(pa + k);
#pragma unroll
      for (int t = 0; t < 4; ++t) bfA[t] = load_frag16(pw[t] + k);
#pragma unroll
      for (int t = 0; t < 4; ++t)
        acc[t] = __builtin_amdgcn_wmma_f32_16x16x32_f16(false, afB, false, bfB[t],
                                                        (short)0, acc[t],
                                                        false, false);
      k += 32;
    }
    afB = load_frag16(pa + k);
#pragma unroll
    for (int t = 0; t < 4; ++t) bfB[t] = load_frag16(pw[t] + k);
#pragma unroll
    for (int t = 0; t < 4; ++t)
      acc[t] = __builtin_amdgcn_wmma_f32_16x16x32_f16(false, afA, false, bfA[t],
                                                      (short)0, acc[t],
                                                      false, false);
#pragma unroll
    for (int t = 0; t < 4; ++t)
      acc[t] = __builtin_amdgcn_wmma_f32_16x16x32_f16(false, afB, false, bfB[t],
                                                      (short)0, acc[t],
                                                      false, false);
  }

#pragma unroll
  for (int t = 0; t < 4; ++t) {
    const int cB = cB0 + t * 16;
    const float bv = bias[cB];
#pragma unroll
    for (int j = 0; j < 8; ++j) {
      int row = m0 + sel * 8 + j;
      float v = acc[t][j] + bv;
      v = v / (1.f + __expf(-v));             // SiLU
      out[(size_t)row * EE + cB] = (_Float16)v;
    }
  }
}

// ---------------------------------------------------------------------------
// LayerNorm over D=768, one block per row, result stored f16 for WMMA.
// ---------------------------------------------------------------------------
__global__ __launch_bounds__(256)
void layernorm_kernel(const float* __restrict__ t,
                      const float* __restrict__ g,
                      const float* __restrict__ bbias,
                      _Float16* __restrict__ xn, int Dd)
{
  const int row = blockIdx.x;
  const float* x = t + (size_t)row * Dd;
  float s = 0.f, s2 = 0.f;
  for (int i = threadIdx.x; i < Dd; i += blockDim.x) {
    float v = x[i]; s += v; s2 += v * v;
  }
  __shared__ float red0[8], red1[8];
  for (int off = 16; off > 0; off >>= 1) {
    s  += __shfl_down(s,  off, 32);
    s2 += __shfl_down(s2, off, 32);
  }
  const int lane = threadIdx.x & 31, w = threadIdx.x >> 5;
  if (lane == 0) { red0[w] = s; red1[w] = s2; }
  __syncthreads();
  if (w == 0) {
    s  = (lane < 8) ? red0[lane] : 0.f;
    s2 = (lane < 8) ? red1[lane] : 0.f;
    for (int off = 4; off > 0; off >>= 1) {
      s  += __shfl_down(s,  off, 32);
      s2 += __shfl_down(s2, off, 32);
    }
    if (lane == 0) { red0[0] = s; red1[0] = s2; }
  }
  __syncthreads();
  const float mu  = red0[0] / Dd;
  const float var = red1[0] / Dd - mu * mu;
  const float rs  = rsqrtf(var + 1e-5f);
  for (int i = threadIdx.x; i < Dd; i += blockDim.x) {
    float v = (x[i] - mu) * rs * g[i] + bbias[i];
    xn[(size_t)row * Dd + i] = (_Float16)v;
  }
}

// ---------------------------------------------------------------------------
// Selective scan. lane = (e-half, n): each 16-lane group owns the N=16 state
// of one channel e; y reduced with __shfl_xor inside the group (wave32).
// ---------------------------------------------------------------------------
__global__ __launch_bounds__(256)
void scan_kernel(const _Float16* __restrict__ xp,    // [B*M][E]
                 const float* __restrict__ Delta,    // [B*M][E]
                 const float* __restrict__ Bp,       // [B*M][N]
                 const float* __restrict__ Cp,       // [B*M][N]
                 const float* __restrict__ Abase,    // [E][N]
                 float* __restrict__ y,              // [B*M][E]
                 int reverse, int accumulate)
{
  const int lane = threadIdx.x & 31;
  const int wave = threadIdx.x >> 5;
  const int n    = lane & 15;
  const int half = lane >> 4;
  const int e    = (blockIdx.x * (blockDim.x >> 5) + wave) * 2 + half;
  const int b    = blockIdx.y;
  const float a  = Abase[(size_t)e * NN + n];
  const size_t rowBase = (size_t)b * MM;
  float h = 0.f;
  for (int step = 0; step < MM; ++step) {
    const int m = reverse ? (MM - 1 - step) : step;
    const size_t r = rowBase + m;
    const float d  = Delta[r * EE + e];
    const float x  = (float)xp[r * EE + e];
    const float Bv = Bp[r * NN + n];
    const float Cv = Cp[r * NN + n];
    h = d * a * h + (d * x) * Bv;
    float c = h * Cv;
    c += __shfl_xor(c, 1, 16);
    c += __shfl_xor(c, 2, 16);
    c += __shfl_xor(c, 4, 16);
    c += __shfl_xor(c, 8, 16);
    if (n == 0) {
      if (accumulate) y[r * EE + e] += c;
      else            y[r * EE + e]  = c;
    }
  }
}

// ---------------------------------------------------------------------------
// Elementwise helpers
// ---------------------------------------------------------------------------
__global__ void gate_kernel(const float* __restrict__ y,
                            const float* __restrict__ sz,
                            _Float16* __restrict__ out, size_t n)
{
  size_t i = (size_t)blockIdx.x * blockDim.x + threadIdx.x;
  if (i < n) out[i] = (_Float16)(y[i] * sz[i]);
}

__global__ void cvt_f32_f16(const float* __restrict__ in,
                            _Float16* __restrict__ out, size_t n)
{
  size_t i = (size_t)blockIdx.x * blockDim.x + threadIdx.x;
  if (i < n) out[i] = (_Float16)in[i];
}

// Wc (E_out, E_in, K) f32 -> Wt (K, E_out, E_in) f16 (contiguous in E_in)
__global__ void cvt_conv_w(const float* __restrict__ in,
                           _Float16* __restrict__ out)
{
  size_t i = (size_t)blockIdx.x * blockDim.x + threadIdx.x;  // over E*E*K
  if (i >= (size_t)EE * EE * KK) return;
  int k  = (int)(i % KK);
  size_t oi = i / KK;                 // o*E + icol
  out[(size_t)k * EE * EE + oi] = (_Float16)in[i];
}

// Zero the halo rows of the padded activation buffer [B][M+2][E].
__global__ void zero_halo_kernel(_Float16* __restrict__ Xpad)
{
  size_t i = (size_t)blockIdx.x * blockDim.x + threadIdx.x;  // over B*2*E
  if (i >= (size_t)BB * 2 * EE) return;
  int b    = (int)(i / (2 * EE));
  int rem  = (int)(i % (2 * EE));
  int side = rem / EE;                // 0 = front halo, 1 = back halo
  int c    = rem % EE;
  size_t row = (size_t)b * MMP + (side ? (MMP - 1) : 0);
  Xpad[row * EE + c] = (_Float16)0.f;
}

// ---------------------------------------------------------------------------
// Host orchestration
// ---------------------------------------------------------------------------
extern "C" void kernel_launch(void* const* d_in, const int* in_sizes, int n_in,
                              void* d_out, int out_size, void* d_ws, size_t ws_size,
                              hipStream_t stream)
{
  (void)in_sizes; (void)n_in; (void)out_size; (void)ws_size;
  const float* t    = (const float*)d_in[0];
  const float* ln_g = (const float*)d_in[1];
  const float* ln_b = (const float*)d_in[2];
  const float* Wx   = (const float*)d_in[3];
  const float* bx   = (const float*)d_in[4];
  const float* Wz   = (const float*)d_in[5];
  const float* bz   = (const float*)d_in[6];
  const float* Wcf  = (const float*)d_in[7];
  const float* bcf  = (const float*)d_in[8];
  const float* Wcb  = (const float*)d_in[9];
  const float* bcb  = (const float*)d_in[10];
  const float* WBf  = (const float*)d_in[11];
  const float* bBf  = (const float*)d_in[12];
  const float* WCf  = (const float*)d_in[13];
  const float* bCf  = (const float*)d_in[14];
  const float* WDf  = (const float*)d_in[15];
  const float* bDf  = (const float*)d_in[16];
  const float* WBb  = (const float*)d_in[17];
  const float* bBb  = (const float*)d_in[18];
  const float* WCb  = (const float*)d_in[19];
  const float* bCb  = (const float*)d_in[20];
  const float* WDb  = (const float*)d_in[21];
  const float* bDb  = (const float*)d_in[22];
  const float* Abf  = (const float*)d_in[23];
  const float* Abb  = (const float*)d_in[24];
  const float* dbf  = (const float*)d_in[25];
  const float* dbb  = (const float*)d_in[26];
  const float* WT   = (const float*)d_in[27];
  const float* bT   = (const float*)d_in[28];

  char* ws = (char*)d_ws;
  size_t off = 0;
  auto alloc = [&](size_t bytes) -> void* {
    void* p = ws + off;
    off = (off + bytes + 255) & ~(size_t)255;
    return p;
  };

  const size_t ROWS  = (size_t)BB * MM;           // 16384
  const size_t ROWSP = (size_t)BB * MMP;          // padded rows
  _Float16* xn     = (_Float16*)alloc(ROWS * DD * 2);
  _Float16* wx_h   = (_Float16*)alloc((size_t)EE * DD * 2);
  _Float16* wz_h   = (_Float16*)alloc((size_t)EE * DD * 2);
  _Float16* wdf_h  = (_Float16*)alloc((size_t)EE * EE * 2);
  _Float16* wdb_h  = (_Float16*)alloc((size_t)EE * EE * 2);
  _Float16* wt_h   = (_Float16*)alloc((size_t)DD * EE * 2);
  _Float16* wbf_h  = (_Float16*)alloc((size_t)NN * EE * 2);
  _Float16* wcfp_h = (_Float16*)alloc((size_t)NN * EE * 2);
  _Float16* wbb_h  = (_Float16*)alloc((size_t)NN * EE * 2);
  _Float16* wcbp_h = (_Float16*)alloc((size_t)NN * EE * 2);
  _Float16* wcf_t  = (_Float16*)alloc((size_t)KK * EE * EE * 2);
  _Float16* wcb_t  = (_Float16*)alloc((size_t)KK * EE * EE * 2);
  _Float16* xprojP = (_Float16*)alloc(ROWSP * EE * 2);   // padded, zero halos
  float*    sz     = (float*)   alloc(ROWS * EE * 4);
  _Float16* xpf    = (_Float16*)alloc(ROWS * EE * 2);
  _Float16* xpb    = (_Float16*)alloc(ROWS * EE * 2);
  float*    delta  = (float*)   alloc(ROWS * EE * 4);
  float*    Bpw    = (float*)   alloc(ROWS * NN * 4);
  float*    Cpw    = (float*)   alloc(ROWS * NN * 4);
  float*    ybuf   = (float*)   alloc(ROWS * EE * 4);
  _Float16* yg     = (_Float16*)alloc(ROWS * EE * 2);

  // 1) LayerNorm -> xn (f16); zero conv halos (workspace is poisoned by harness)
  layernorm_kernel<<<(unsigned)ROWS, 256, 0, stream>>>(t, ln_g, ln_b, xn, DD);
  {
    size_t n = (size_t)BB * 2 * EE;
    zero_halo_kernel<<<(unsigned)((n + 255) / 256), 256, 0, stream>>>(xprojP);
  }

  // 2) Weight conversion to f16
  auto cvt = [&](const float* src, _Float16* dst, size_t n) {
    cvt_f32_f16<<<(unsigned)((n + 255) / 256), 256, 0, stream>>>(src, dst, n);
  };
  cvt(Wx,  wx_h,   (size_t)EE * DD);
  cvt(Wz,  wz_h,   (size_t)EE * DD);
  cvt(WDf, wdf_h,  (size_t)EE * EE);
  cvt(WDb, wdb_h,  (size_t)EE * EE);
  cvt(WT,  wt_h,   (size_t)DD * EE);
  cvt(WBf, wbf_h,  (size_t)NN * EE);
  cvt(WCf, wcfp_h, (size_t)NN * EE);
  cvt(WBb, wbb_h,  (size_t)NN * EE);
  cvt(WCb, wcbp_h, (size_t)NN * EE);
  {
    size_t n = (size_t)EE * EE * KK;
    cvt_conv_w<<<(unsigned)((n + 255) / 256), 256, 0, stream>>>(Wcf, wcf_t);
    cvt_conv_w<<<(unsigned)((n + 255) / 256), 256, 0, stream>>>(Wcb, wcb_t);
  }

  dim3 blk(128);
  dim3 gE4(EE / 64, (unsigned)(ROWS / 64));   // (24, 256) 16x64 wave tiles
  dim3 gN1(1,       (unsigned)(ROWS / 64));   // N=16 projections
  dim3 gD4(DD / 64, (unsigned)(ROWS / 64));   // (12, 256)

  // 3) x_proj (f16, padded layout) and silu(z) (f32)
  wmma_gemm_kernel<4, ACT_NONE, true, false, true><<<gE4, blk, 0, stream>>>(
      xn, DD, wx_h, DD, bx, nullptr, nullptr, nullptr, xprojP, EE, DD);
  wmma_gemm_kernel<4, ACT_SILU, false, false, false><<<gE4, blk, 0, stream>>>(
      xn, DD, wz_h, DD, bz, nullptr, nullptr, sz, nullptr, EE, DD);

  // 4) conv + silu, both directions (maskless over zero-halo input)
  wmma_conv_silu_kernel<<<gE4, blk, 0, stream>>>(xprojP, wcf_t, bcf, xpf);
  wmma_conv_silu_kernel<<<gE4, blk, 0, stream>>>(xprojP, wcb_t, bcb, xpb);

  // 5) forward: B, C, Delta projections then scan
  wmma_gemm_kernel<1, ACT_NONE, false, false, false><<<gN1, blk, 0, stream>>>(
      xpf, EE, wbf_h, EE, bBf, nullptr, nullptr, Bpw, nullptr, NN, EE);
  wmma_gemm_kernel<1, ACT_NONE, false, false, false><<<gN1, blk, 0, stream>>>(
      xpf, EE, wcfp_h, EE, bCf, nullptr, nullptr, Cpw, nullptr, NN, EE);
  wmma_gemm_kernel<4, ACT_SOFTPLUS, false, false, false><<<gE4, blk, 0, stream>>>(
      xpf, EE, wdf_h, EE, bDf, dbf, nullptr, delta, nullptr, EE, EE);
  scan_kernel<<<dim3(EE / 16, BB), 256, 0, stream>>>(
      xpf, delta, Bpw, Cpw, Abf, ybuf, /*reverse=*/0, /*accumulate=*/0);

  // 6) backward: reuse Delta/B/C buffers, scan in reverse, accumulate into y
  wmma_gemm_kernel<1, ACT_NONE, false, false, false><<<gN1, blk, 0, stream>>>(
      xpb, EE, wbb_h, EE, bBb, nullptr, nullptr, Bpw, nullptr, NN, EE);
  wmma_gemm_kernel<1, ACT_NONE, false, false, false><<<gN1, blk, 0, stream>>>(
      xpb, EE, wcbp_h, EE, bCb, nullptr, nullptr, Cpw, nullptr, NN, EE);
  wmma_gemm_kernel<4, ACT_SOFTPLUS, false, false, false><<<gE4, blk, 0, stream>>>(
      xpb, EE, wdb_h, EE, bDb, dbb, nullptr, delta, nullptr, EE, EE);
  scan_kernel<<<dim3(EE / 16, BB), 256, 0, stream>>>(
      xpb, delta, Bpw, Cpw, Abb, ybuf, /*reverse=*/1, /*accumulate=*/1);

  // 7) gate with silu(z), convert to f16 for the final WMMA GEMM
  {
    size_t n = ROWS * EE;
    gate_kernel<<<(unsigned)((n + 255) / 256), 256, 0, stream>>>(ybuf, sz, yg, n);
  }

  // 8) output projection + bias + residual -> d_out (f32)
  wmma_gemm_kernel<4, ACT_NONE, false, true, false><<<gD4, blk, 0, stream>>>(
      yg, EE, wt_h, EE, bT, nullptr, t, (float*)d_out, nullptr, DD, EE);
}
